// GNN_90589450207317
// MI455X (gfx1250) — matrix operand order repaired and checked by reading, not compiled, not run on hardware
//
#include <hip/hip_runtime.h>
#include <hip/hip_bf16.h>

#define NN  100000
#define NE  1600000
#define IND 128
#define HD  64

typedef __attribute__((ext_vector_type(16))) __bf16 v16bf;
typedef __attribute__((ext_vector_type(8)))  float  v8f;

union V16U { uint4 q[2]; v16bf v; };

__device__ __forceinline__ v16bf ld2(const __bf16* p0, const __bf16* p1) {
  V16U u;
  u.q[0] = *(const uint4*)p0;
  u.q[1] = *(const uint4*)p1;
  return u.v;
}

// ---------------- degree / dinv ----------------
__global__ void k_zero_u32(unsigned* __restrict__ p, int n) {
  int i = blockIdx.x * 256 + threadIdx.x;
  if (i < n) p[i] = 0u;
}

__global__ void k_degree(const int* __restrict__ dst, unsigned* __restrict__ deg) {
  int e = blockIdx.x * 256 + threadIdx.x;
  if (e < NE) atomicAdd(&deg[dst[e]], 1u);
}

__global__ void k_dinv(const unsigned* __restrict__ deg, float* __restrict__ dinv, int n) {
  int i = blockIdx.x * 256 + threadIdx.x;
  if (i < n) dinv[i] = rsqrtf((float)deg[i] + 1.0f);
}

// ------------- fp32 -> bf16 hi/lo split (optional fused ReLU) -------------
__global__ void k_split(const float* __restrict__ in, __bf16* __restrict__ hi,
                        __bf16* __restrict__ lo, int n, int relu) {
  int i = blockIdx.x * 256 + threadIdx.x;
  if (i >= n) return;
  float v = in[i];
  if (relu) v = fmaxf(v, 0.0f);
  __bf16 h = (__bf16)v;
  hi[i] = h;
  lo[i] = (__bf16)(v - (float)h);
}

// ------------- WMMA GEMM: [nrows x K] @ [K x 64] -------------
// MODE 0 (pre layer):   out = bf16_split(acc + bias)            -> Ohi/Olo
// MODE 1 (GCN layer):   HW = acc ; AGG = acc*dinv[row]^2 + bias -> HW/AGG
template <int MODE>
__global__ __launch_bounds__(256) void k_gemm(
    const __bf16* __restrict__ Ahi, const __bf16* __restrict__ Alo,
    const __bf16* __restrict__ Bhi, const __bf16* __restrict__ Blo,
    const float* __restrict__ bias, const float* __restrict__ dinv,
    int nrows, int K,
    float* __restrict__ HW, float* __restrict__ AGG,
    __bf16* __restrict__ Ohi, __bf16* __restrict__ Olo)
{
  const int lane = threadIdx.x & 31;
  const int wv   = threadIdx.x >> 5;
  const int rowBase = blockIdx.x * 128 + wv * 16;
  if (rowBase >= nrows) return;            // wave-uniform; EXEC stays all-1s
  const int mm = lane & 15;                // A: M index / C: N index
  const int hf = lane >> 4;                // A/C half select
  int arow = rowBase + mm;
  if (arow > nrows - 1) arow = nrows - 1;  // clamp instead of masking loads
  const __bf16* aH = Ahi + (size_t)arow * K;
  const __bf16* aL = Alo + (size_t)arow * K;

  v8f acc[4] = {};
  for (int kb = 0; kb < K; kb += 32) {
    const int e0 = kb + hf * 8;            // 16-bit A layout: K 0..7 / 16..23 (+8 for half 1)
    v16bf ah = ld2(aH + e0, aH + e0 + 16);
    v16bf al = ld2(aL + e0, aL + e0 + 16);
    const __bf16* bH = Bhi + (size_t)(kb + hf * 8) * HD + mm;  // B layout: lane%16 == N col, K order as A
    const __bf16* bL = Blo + (size_t)(kb + hf * 8) * HD + mm;
#pragma unroll
    for (int t = 0; t < 4; ++t) {
      v16bf bh, bl;
#pragma unroll
      for (int s = 0; s < 16; ++s) {
        const int o = (s < 8 ? s : s + 8) * HD + t * 16;
        bh[s] = bH[o];
        bl[s] = bL[o];
      }
      acc[t] = __builtin_amdgcn_wmma_f32_16x16x32_bf16(false, ah, false, bh, (short)0, acc[t], false, false);
      acc[t] = __builtin_amdgcn_wmma_f32_16x16x32_bf16(false, ah, false, bl, (short)0, acc[t], false, false);
      acc[t] = __builtin_amdgcn_wmma_f32_16x16x32_bf16(false, al, false, bh, (short)0, acc[t], false, false);
    }
  }

  float bt[4];
#pragma unroll
  for (int t = 0; t < 4; ++t) bt[t] = bias[t * 16 + mm];

#pragma unroll
  for (int r = 0; r < 8; ++r) {
    const int rout = rowBase + hf * 8 + r;   // C layout: VGPR r -> M = r (+8 for lanes>=16)
    if (rout >= nrows) continue;
    float d2 = 0.0f;
    if (MODE == 1) { float dv = dinv[rout]; d2 = dv * dv; }
#pragma unroll
    for (int t = 0; t < 4; ++t) {
      const int n = t * 16 + mm;
      const float v = acc[t][r];
      if (MODE == 0) {
        float s = v + bt[t];
        __bf16 h = (__bf16)s;
        Ohi[(size_t)rout * HD + n] = h;
        Olo[(size_t)rout * HD + n] = (__bf16)(s - (float)h);
      } else {
        HW[(size_t)rout * HD + n]  = v;
        AGG[(size_t)rout * HD + n] = v * d2 + bt[t];   // self-loop term + bias
      }
    }
  }
}

// ------------- edge scatter: agg[dst] += hw[src]*dinv[src]*dinv[dst] -------------
// one edge per wave32; lane handles 2 channels; hw/agg are L2-resident (25.6 MB each)
__global__ __launch_bounds__(256) void k_scatter(const int* __restrict__ src,
                                                 const int* __restrict__ dst,
                                                 const float* __restrict__ dinv,
                                                 const float* __restrict__ hw,
                                                 float* __restrict__ agg)
{
  const int lane = threadIdx.x & 31;
  const int e = blockIdx.x * 8 + (threadIdx.x >> 5);
  if (e >= NE) return;
  const int s = src[e];
  const int d = dst[e];
  const float nrm = dinv[s] * dinv[d];
  const float2 v = ((const float2*)(hw + (size_t)s * HD))[lane];
  float* ap = agg + (size_t)d * HD + lane * 2;
  unsafeAtomicAdd(ap,     v.x * nrm);  // native global_atomic_add_f32, no CAS loop
  unsafeAtomicAdd(ap + 1, v.y * nrm);
}

// ------------- row L2-normalize -------------
__global__ __launch_bounds__(256) void k_norm(const float* __restrict__ agg,
                                              float* __restrict__ out)
{
  const int lane = threadIdx.x & 31;
  const int node = blockIdx.x * 8 + (threadIdx.x >> 5);
  if (node >= NN) return;
  const float2 v = ((const float2*)(agg + (size_t)node * HD))[lane];
  float s = v.x * v.x + v.y * v.y;
#pragma unroll
  for (int m = 16; m >= 1; m >>= 1) s += __shfl_xor(s, m, 32);
  const float scale = 1.0f / fmaxf(sqrtf(s), 1e-12f);
  float2 o;
  o.x = v.x * scale;
  o.y = v.y * scale;
  ((float2*)(out + (size_t)node * HD))[lane] = o;
}

// ---------------- launcher ----------------
extern "C" void kernel_launch(void* const* d_in, const int* in_sizes, int n_in,
                              void* d_out, int out_size, void* d_ws, size_t ws_size,
                              hipStream_t stream) {
  const float* x  = (const float*)d_in[0];
  const int*   ei = (const int*)d_in[1];        // [2, NE]: row0 = src, row1 = dst
  const float* Wp = (const float*)d_in[2];
  const float* bp = (const float*)d_in[3];
  const float* W1 = (const float*)d_in[4];
  const float* b1 = (const float*)d_in[5];
  const float* W2 = (const float*)d_in[6];
  const float* b2 = (const float*)d_in[7];
  const float* W3 = (const float*)d_in[8];
  const float* b3 = (const float*)d_in[9];
  float* out = (float*)d_out;

  char* ws = (char*)d_ws;
  size_t off = 0;
  auto alloc = [&](size_t bytes) -> char* {
    off = (off + 255) & ~(size_t)255;
    char* p = ws + off;
    off += bytes;
    return p;
  };

  unsigned* deg  = (unsigned*)alloc((size_t)NN * 4);
  float*    dinv = (float*)alloc((size_t)NN * 4);
  // 51.2 MB region: {x_hi, x_lo} for the pre-GEMM, then reused as {HW, AGG}
  char* big = alloc((size_t)NN * 512);
  __bf16* xhi = (__bf16*)big;
  __bf16* xlo = (__bf16*)(big + (size_t)NN * 256);
  float*  HW  = (float*)big;
  float*  AGG = (float*)(big + (size_t)NN * 256);
  __bf16* Hhi = (__bf16*)alloc((size_t)NN * HD * 2);
  __bf16* Hlo = (__bf16*)alloc((size_t)NN * HD * 2);
  __bf16* wphi = (__bf16*)alloc((size_t)IND * HD * 2);
  __bf16* wplo = (__bf16*)alloc((size_t)IND * HD * 2);
  __bf16* w1hi = (__bf16*)alloc((size_t)HD * HD * 2);
  __bf16* w1lo = (__bf16*)alloc((size_t)HD * HD * 2);
  __bf16* w2hi = (__bf16*)alloc((size_t)HD * HD * 2);
  __bf16* w2lo = (__bf16*)alloc((size_t)HD * HD * 2);
  __bf16* w3hi = (__bf16*)alloc((size_t)HD * HD * 2);
  __bf16* w3lo = (__bf16*)alloc((size_t)HD * HD * 2);

  // degrees of A + I -> dinv
  k_zero_u32<<<(NN + 255) / 256, 256, 0, stream>>>(deg, NN);
  k_degree<<<(NE + 255) / 256, 256, 0, stream>>>(ei + NE, deg);
  k_dinv<<<(NN + 255) / 256, 256, 0, stream>>>(deg, dinv, NN);

  // bf16 hi/lo splits of x and all weight matrices
  k_split<<<((NN * IND) + 255) / 256, 256, 0, stream>>>(x, xhi, xlo, NN * IND, 0);
  k_split<<<(IND * HD + 255) / 256, 256, 0, stream>>>(Wp, wphi, wplo, IND * HD, 0);
  k_split<<<(HD * HD + 255) / 256, 256, 0, stream>>>(W1, w1hi, w1lo, HD * HD, 0);
  k_split<<<(HD * HD + 255) / 256, 256, 0, stream>>>(W2, w2hi, w2lo, HD * HD, 0);
  k_split<<<(HD * HD + 255) / 256, 256, 0, stream>>>(W3, w3hi, w3lo, HD * HD, 0);

  const dim3 gg((NN + 127) / 128), gb(256);
  const int scatGrid = (NE + 7) / 8;

  // pre linear: h = x @ W_pre + b_pre  (stored directly as bf16 hi/lo)
  k_gemm<0><<<gg, gb, 0, stream>>>(xhi, xlo, wphi, wplo, bp, nullptr, NN, IND,
                                   nullptr, nullptr, Hhi, Hlo);

  // layer 1
  k_gemm<1><<<gg, gb, 0, stream>>>(Hhi, Hlo, w1hi, w1lo, b1, dinv, NN, HD,
                                   HW, AGG, nullptr, nullptr);
  k_scatter<<<scatGrid, 256, 0, stream>>>(ei, ei + NE, dinv, HW, AGG);
  k_split<<<((NN * HD) + 255) / 256, 256, 0, stream>>>(AGG, Hhi, Hlo, NN * HD, 1); // ReLU fused

  // layer 2
  k_gemm<1><<<gg, gb, 0, stream>>>(Hhi, Hlo, w2hi, w2lo, b2, dinv, NN, HD,
                                   HW, AGG, nullptr, nullptr);
  k_scatter<<<scatGrid, 256, 0, stream>>>(ei, ei + NE, dinv, HW, AGG);
  k_split<<<((NN * HD) + 255) / 256, 256, 0, stream>>>(AGG, Hhi, Hlo, NN * HD, 1); // ReLU fused

  // layer 3 (no ReLU)
  k_gemm<1><<<gg, gb, 0, stream>>>(Hhi, Hlo, w3hi, w3lo, b3, dinv, NN, HD,
                                   HW, AGG, nullptr, nullptr);
  k_scatter<<<scatGrid, 256, 0, stream>>>(ei, ei + NE, dinv, HW, AGG);

  // row-wise L2 normalize -> output
  k_norm<<<(NN + 7) / 8, 256, 0, stream>>>(AGG, out);
}
